// window_attenblock_28810640622052
// MI455X (gfx1250) — matrix operand order repaired and checked by reading, not compiled, not run on hardware
//
#include <hip/hip_runtime.h>

typedef __attribute__((ext_vector_type(16))) __bf16 v16bf;
typedef __attribute__((ext_vector_type(8)))  float  v8f;
typedef __attribute__((ext_vector_type(4)))  unsigned int u32x4;
typedef __attribute__((ext_vector_type(2)))  unsigned int u32x2;
typedef __attribute__((ext_vector_type(4)))  float  f32x4;
typedef __attribute__((ext_vector_type(8)))  int    i32x8;
typedef __attribute__((ext_vector_type(4)))  int    i32x4;
typedef unsigned int u32;
typedef unsigned short u16;
typedef unsigned long long u64;

union Frag16 { v16bf v; u32x4 q[2]; };

__device__ __forceinline__ u16 f2bf(float f) {
    u32 u = __builtin_bit_cast(u32, f);
    u32 r = u + 0x7FFFu + ((u >> 16) & 1u);   // round-to-nearest-even
    return (u16)(r >> 16);
}

__device__ __forceinline__ v8f wmma_bf16(const Frag16& a, const Frag16& b, v8f c) {
    return __builtin_amdgcn_wmma_f32_16x16x32_bf16(
        false, a.v, false, b.v, (short)0, c, false, false);
}

// ---- LDS byte-offset of a __shared__ object (generic -> AS(3) -> int) ----
typedef __attribute__((address_space(3))) const char lds_cchar;
__device__ __forceinline__ u32 lds_addr_of(const void* p) {
    return (u32)(uintptr_t)(lds_cchar*)p;
}

// ---- async / tensor counters ----
__device__ __forceinline__ void wait_async0() {
#if __has_builtin(__builtin_amdgcn_s_wait_asynccnt)
    __builtin_amdgcn_s_wait_asynccnt(0);
#else
    asm volatile("s_wait_asynccnt 0x0" ::: "memory");
#endif
}
__device__ __forceinline__ void async_copy_b64_to_lds(u32 lds_byte, const void* gaddr) {
    asm volatile("global_load_async_to_lds_b64 %0, %1, off"
                 :: "v"(lds_byte), "v"((u64)(uintptr_t)gaddr) : "memory");
}

// ---- TDM: load a windowed 3D tile (32 bf16 x 8 tokens x 8 rows) into LDS ----
// LDS layout produced: 64 rows x 32 halfs with 16B padding per row (pitch 40).
#define HAS_TDM __has_builtin(__builtin_amdgcn_tensor_load_to_lds)
#if HAS_TDM
__device__ __forceinline__ void tdm_load_window(const u16* gsrc, u32 lds_byte) {
    const u64 ga = (u64)(uintptr_t)gsrc;
    // D# group 0: count=1 | lds_addr | global_addr(57b) | type=2
    u32x4 g0 = { 1u, lds_byte, (u32)ga,
                 (u32)((ga >> 32) & 0x01FFFFFFu) | (2u << 30) };
    // D# group 1: data_size=1(2B), pad_enable, pad_interval=16DW(code 3),
    // pad_amount=4DW(code 3); tensor_dim0=384, tensor_dim1=64;
    // tile = 32 x 8 x 8; stride0=384, stride1=24576.
    i32x8 g1 = { (int)((1u << 16) | (1u << 20) | (3u << 22) | (3u << 25)),
                 (int)(384u << 16),               // tensor_dim0[15:0] @ [31:16]
                 (int)((384u >> 16) | (64u << 16)),   // dim0 hi | dim1 lo
                 (int)(32u << 16),                // dim1 hi=0 | tile_dim0=32
                 (int)(8u | (8u << 16)),          // tile_dim1=8 | tile_dim2=8
                 (int)384,                        // dim0_stride[31:0]
                 (int)(24576u << 16),             // dim0_str hi=0 | dim1_str lo16
                 (int)(24576u >> 16) };           // dim1_stride[47:16]
    i32x4 g2 = { 512, 0, 0, 0 };                  // tensor_dim2=512
    i32x4 g3 = { 0, 0, 0, 0 };
#if defined(__clang_major__) && (__clang_major__ >= 23)
    i32x8 g4 = { 0, 0, 0, 0, 0, 0, 0, 0 };
    __builtin_amdgcn_tensor_load_to_lds(g0, g1, g2, g3, g4, 0);
#else
    __builtin_amdgcn_tensor_load_to_lds(g0, g1, g2, g3, 0);
#endif
}
#endif

// ---- DPP16 butterfly reductions across a 16-lane half (pure VALU, no LDS) ----
template<int CTRL>
__device__ __forceinline__ float dppf(float x) {
    int i = __builtin_bit_cast(int, x);
    return __builtin_bit_cast(float,
        __builtin_amdgcn_update_dpp(i, i, CTRL, 0xF, 0xF, true));
}
__device__ __forceinline__ float rowmax16(float x) {
    x = fmaxf(x, dppf<0xB1>(x));    // quad_perm [1,0,3,2]  (xor 1)
    x = fmaxf(x, dppf<0x4E>(x));    // quad_perm [2,3,0,1]  (xor 2)
    x = fmaxf(x, dppf<0x141>(x));   // row_half_mirror      (xor 4 effect)
    x = fmaxf(x, dppf<0x140>(x));   // row_mirror           (xor 8 effect)
    return x;
}
__device__ __forceinline__ float rowsum16(float x) {
    x += dppf<0xB1>(x);
    x += dppf<0x4E>(x);
    x += dppf<0x141>(x);
    x += dppf<0x140>(x);
    return x;
}

// ---------------------------------------------------------------------------
// GEMM:  Y[M=32768, N=384] = X[32768,384] @ W^T + bias, optionally * scale
// Block: 256 threads = 8 waves (2 M-waves x 4 N-waves).
// Block tile 64M x 192N; wave tile 32M x 48N = 2x3 WMMA accumulators.
// ---------------------------------------------------------------------------
constexpr int NC  = 384;
constexpr int BMB = 64, BNB = 192, BK = 32;
constexpr int PIT = 40;                 // LDS pitch in halfs, rows 16B-aligned

template<bool XBF16, bool OUTBF16>
__global__ __launch_bounds__(256) void gemm_kernel(
    const void* __restrict__ Xv, const float* __restrict__ Wt,
    const float* __restrict__ bias, void* __restrict__ Yv, float scale)
{
    __shared__ u16 As[BMB * PIT];
    __shared__ u16 Bs[BNB * PIT];

    const int tid  = threadIdx.x;
    const int lane = tid & 31;
    const int wave = tid >> 5;
    const int wm   = wave & 1;
    const int wn   = wave >> 1;
    const int half = lane >> 4;
    const int r    = lane & 15;
    const int m0   = blockIdx.y * BMB;
    const int n0   = blockIdx.x * BNB;

    const int ar = tid >> 3;            // 0..31
    const int ac = (tid & 7) * 4;       // 0..28

    v8f acc[2][3] = {};

    for (int k0 = 0; k0 < NC; k0 += BK) {
        __syncthreads();
        // --- stage A tile: rows m0..m0+63, cols k0..k0+31 ---
        #pragma unroll
        for (int s = 0; s < 2; ++s) {
            const int rr = ar + s * 32;
            if (XBF16) {
                // bf16 activations: raw 8-byte copy -> async LDS-DMA path
                const u16* X = (const u16*)Xv;
                async_copy_b64_to_lds(lds_addr_of(&As[rr * PIT + ac]),
                                      &X[(size_t)(m0 + rr) * NC + k0 + ac]);
            } else {
                const float* X = (const float*)Xv;
                f32x4 f = *(const f32x4*)&X[(size_t)(m0 + rr) * NC + k0 + ac];
                u16 h[4] = { f2bf(f.x), f2bf(f.y), f2bf(f.z), f2bf(f.w) };
                *(u32x2*)&As[rr * PIT + ac] = *(const u32x2*)h;
            }
        }
        // prefetch next X K-tile into cache (global_prefetch path)
        if (!XBF16 && k0 + BK < NC) {
            const float* X = (const float*)Xv;
            __builtin_prefetch(&X[(size_t)(m0 + ar) * NC + k0 + BK + ac], 0, 1);
        }
        // --- stage B tile: W rows n0..n0+191, cols k0..k0+31 ---
        #pragma unroll
        for (int s = 0; s < 6; ++s) {
            const int rr = ar + s * 32;
            f32x4 f = *(const f32x4*)&Wt[(size_t)(n0 + rr) * NC + k0 + ac];
            u16 h[4] = { f2bf(f.x), f2bf(f.y), f2bf(f.z), f2bf(f.w) };
            *(u32x2*)&Bs[rr * PIT + ac] = *(const u32x2*)h;
        }
        if (XBF16) wait_async0();
        __syncthreads();

        Frag16 a[2], b[3];
        #pragma unroll
        for (int mi = 0; mi < 2; ++mi) {
            const int row = wm * 32 + mi * 16 + r;
            a[mi].q[0] = *(const u32x4*)&As[row * PIT + half * 8];
            a[mi].q[1] = *(const u32x4*)&As[row * PIT + 16 + half * 8];
        }
        #pragma unroll
        for (int nj = 0; nj < 3; ++nj) {
            const int row = wn * 48 + nj * 16 + r;
            b[nj].q[0] = *(const u32x4*)&Bs[row * PIT + half * 8];
            b[nj].q[1] = *(const u32x4*)&Bs[row * PIT + 16 + half * 8];
        }
        #pragma unroll
        for (int mi = 0; mi < 2; ++mi)
            #pragma unroll
            for (int nj = 0; nj < 3; ++nj)
                acc[mi][nj] = wmma_bf16(a[mi], b[nj], acc[mi][nj]);
    }

    #pragma unroll
    for (int nj = 0; nj < 3; ++nj) {
        const int n = n0 + wn * 48 + nj * 16 + r;
        const float bn = bias[n];
        #pragma unroll
        for (int mi = 0; mi < 2; ++mi) {
            #pragma unroll
            for (int i = 0; i < 8; ++i) {
                const int m = m0 + wm * 32 + mi * 16 + i + 8 * half;
                float val = (acc[mi][nj][i] + bn) * scale;
                if (OUTBF16) ((u16*)Yv)[(size_t)m * NC + n] = f2bf(val);
                else         ((float*)Yv)[(size_t)m * NC + n] = val;
            }
        }
    }
}

// ---------------------------------------------------------------------------
// bias[h][m][n] = bias_table[rel_index[m*64+n]][h]
// ---------------------------------------------------------------------------
__global__ void bias_kernel(const float* __restrict__ table,
                            const int* __restrict__ rel,
                            float* __restrict__ out)
{
    int idx = blockIdx.x * 256 + threadIdx.x;     // 12*4096 total
    if (idx >= 12 * 4096) return;
    int h  = idx / 4096;
    int mn = idx - h * 4096;
    out[idx] = table[rel[mn] * 12 + h];
}

// ---------------------------------------------------------------------------
// Attention: one block per (window, head). 128 threads = 4 waves; wave i
// owns S rows [16i,16i+16). q/k staged by the Tensor Data Mover (3D tile
// with LDS padding), v^T staged manually. S=q k^T -> softmax (DPP
// reductions) -> +bias AFTER softmax (faithful) -> P @ v.
// ---------------------------------------------------------------------------
__global__ __launch_bounds__(128) void attn_kernel(
    const u16* __restrict__ q, const u16* __restrict__ k,
    const u16* __restrict__ v, const float* __restrict__ biasw,
    u16* __restrict__ o)
{
    __shared__ u16 qs[64 * 40];
    __shared__ u16 ks[64 * 40];
    __shared__ u16 vt[32 * 72];   // v transposed: [d][t]
    __shared__ u16 ps[64 * 72];   // P (post-softmax + bias), bf16

    const int w  = blockIdx.x;              // 0..511
    const int h  = blockIdx.y;              // 0..11
    const int bi = w >> 6, wy = (w >> 3) & 7, wx = w & 7;
    const int base = bi * 4096 + wy * 512 + wx * 8;

    const int tid  = threadIdx.x;
    const int lane = tid & 31;
    const int wave = tid >> 5;
    const int half = lane >> 4;
    const int r    = lane & 15;

    const size_t tile0 = (size_t)base * 384 + (size_t)h * 32;  // tile start (elems)

#if HAS_TDM
    if (wave == 0)      tdm_load_window(&q[tile0], lds_addr_of(qs));
    else if (wave == 1) tdm_load_window(&k[tile0], lds_addr_of(ks));
#endif
    // ---- stage v^T (transpose in registers -> LDS) ----
    {
        const int t  = tid >> 1;            // token 0..63
        const int c0 = (tid & 1) * 16;
        const size_t row = tile0 + (size_t)((t >> 3) * 64 + (t & 7)) * 384 + c0
                         - (size_t)0;       // == gathered row base + c0
        const size_t g = (size_t)(base + (t >> 3) * 64 + (t & 7)) * 384
                       + (size_t)h * 32 + c0;
#if !HAS_TDM
        *(u32x4*)&qs[t * 40 + c0]     = *(const u32x4*)&q[g];
        *(u32x4*)&qs[t * 40 + c0 + 8] = *(const u32x4*)&q[g + 8];
        *(u32x4*)&ks[t * 40 + c0]     = *(const u32x4*)&k[g];
        *(u32x4*)&ks[t * 40 + c0 + 8] = *(const u32x4*)&k[g + 8];
#endif
        (void)row;
        u32x4 d0 = *(const u32x4*)&v[g];
        u32x4 d1 = *(const u32x4*)&v[g + 8];
        #pragma unroll
        for (int j = 0; j < 8; ++j)
            vt[(c0 + j) * 72 + t] = (u16)(d0[j >> 1] >> ((j & 1) * 16));
        #pragma unroll
        for (int j = 0; j < 8; ++j)
            vt[(c0 + 8 + j) * 72 + t] = (u16)(d1[j >> 1] >> ((j & 1) * 16));
    }
#if HAS_TDM
    if (wave < 2) __builtin_amdgcn_s_wait_tensorcnt(0);
#endif
    __syncthreads();

    // ---- S = q k^T  (wave's 16 rows x 64 cols, K = 32) ----
    v8f s[4];
    {
        Frag16 a;
        a.q[0] = *(const u32x4*)&qs[(wave * 16 + r) * 40 + half * 8];
        a.q[1] = *(const u32x4*)&qs[(wave * 16 + r) * 40 + 16 + half * 8];
        #pragma unroll
        for (int j = 0; j < 4; ++j) {
            Frag16 b;
            b.q[0] = *(const u32x4*)&ks[(j * 16 + r) * 40 + half * 8];
            b.q[1] = *(const u32x4*)&ks[(j * 16 + r) * 40 + 16 + half * 8];
            v8f z = {};
            s[j] = wmma_bf16(a, b, z);
        }
    }

    // ---- softmax over each row (DPP16 butterflies), bias added AFTER ----
    #pragma unroll
    for (int vi = 0; vi < 8; ++vi) {
        float mx = fmaxf(fmaxf(s[0][vi], s[1][vi]), fmaxf(s[2][vi], s[3][vi]));
        mx = rowmax16(mx);
        float e[4], sum = 0.f;
        #pragma unroll
        for (int j = 0; j < 4; ++j) { e[j] = __expf(s[j][vi] - mx); sum += e[j]; }
        sum = rowsum16(sum);
        const float inv = 1.0f / sum;
        const int m = wave * 16 + vi + half * 8;
        const float* bp = &biasw[((size_t)h * 64 + m) * 64];
        #pragma unroll
        for (int j = 0; j < 4; ++j) {
            float p = e[j] * inv + bp[j * 16 + r];
            ps[m * 72 + j * 16 + r] = f2bf(p);
        }
    }
    __syncthreads();

    // ---- O = P @ v  (K = 64 in 2 steps, N = 32 in 2 tiles) ----
    v8f oacc[2] = { {}, {} };
    #pragma unroll
    for (int kk = 0; kk < 2; ++kk) {
        Frag16 pa;
        pa.q[0] = *(const u32x4*)&ps[(wave * 16 + r) * 72 + kk * 32 + half * 8];
        pa.q[1] = *(const u32x4*)&ps[(wave * 16 + r) * 72 + kk * 32 + 16 + half * 8];
        #pragma unroll
        for (int jn = 0; jn < 2; ++jn) {
            Frag16 vb;
            vb.q[0] = *(const u32x4*)&vt[(jn * 16 + r) * 72 + kk * 32 + half * 8];
            vb.q[1] = *(const u32x4*)&vt[(jn * 16 + r) * 72 + kk * 32 + 16 + half * 8];
            oacc[jn] = wmma_bf16(pa, vb, oacc[jn]);
        }
    }

    // ---- scatter O back to token layout (bf16 workspace) ----
    #pragma unroll
    for (int vi = 0; vi < 8; ++vi) {
        const int t = wave * 16 + vi + half * 8;
        const size_t row = (size_t)(base + (t >> 3) * 64 + (t & 7)) * 384
                         + (size_t)h * 32;
        #pragma unroll
        for (int jn = 0; jn < 2; ++jn)
            o[row + jn * 16 + r] = f2bf(oacc[jn][vi]);
    }
}

// ---------------------------------------------------------------------------
extern "C" void kernel_launch(void* const* d_in, const int* in_sizes, int n_in,
                              void* d_out, int out_size, void* d_ws, size_t ws_size,
                              hipStream_t stream)
{
    const float* x1  = (const float*)d_in[0];
    const float* x2  = (const float*)d_in[1];
    const float* x3  = (const float*)d_in[2];
    const float* x4  = (const float*)d_in[3];
    const float* wq1 = (const float*)d_in[4];  const float* bq1 = (const float*)d_in[5];
    const float* wq2 = (const float*)d_in[6];  const float* bq2 = (const float*)d_in[7];
    const float* wq3 = (const float*)d_in[8];  const float* bq3 = (const float*)d_in[9];
    const float* wk  = (const float*)d_in[10]; const float* bk  = (const float*)d_in[11];
    const float* wv  = (const float*)d_in[12]; const float* bv  = (const float*)d_in[13];
    const float* wp1 = (const float*)d_in[14]; const float* bp1 = (const float*)d_in[15];
    const float* wp2 = (const float*)d_in[16]; const float* bp2 = (const float*)d_in[17];
    const float* wp3 = (const float*)d_in[18]; const float* bp3 = (const float*)d_in[19];
    const float* btab = (const float*)d_in[20];
    const int*   ridx = (const int*)d_in[21];

    const size_t S1 = (size_t)32768 * 384 * sizeof(u16);   // one bf16 activation buf
    char* ws = (char*)d_ws;
    u16* q1b = (u16*)(ws + 0 * S1);
    u16* q2b = (u16*)(ws + 1 * S1);
    u16* q3b = (u16*)(ws + 2 * S1);
    u16* kb  = (u16*)(ws + 3 * S1);
    u16* vb  = (u16*)(ws + 4 * S1);
    u16* o1b = (u16*)(ws + 5 * S1);
    u16* o2b = (u16*)(ws + 6 * S1);
    u16* o3b = (u16*)(ws + 7 * S1);
    float* biasw = (float*)(ws + 8 * S1);

    const float scale = 0.17677669529663687f;   // 32^-0.5

    bias_kernel<<<192, 256, 0, stream>>>(btab, ridx, biasw);

    dim3 gg(NC / BNB, 32768 / BMB);   // (2, 512)
    gemm_kernel<false, true><<<gg, 256, 0, stream>>>(x2, wq1, bq1, q1b, scale);
    gemm_kernel<false, true><<<gg, 256, 0, stream>>>(x3, wq2, bq2, q2b, scale);
    gemm_kernel<false, true><<<gg, 256, 0, stream>>>(x4, wq3, bq3, q3b, scale);
    gemm_kernel<false, true><<<gg, 256, 0, stream>>>(x1, wk,  bk,  kb,  1.0f);
    gemm_kernel<false, true><<<gg, 256, 0, stream>>>(x1, wv,  bv,  vb,  1.0f);

    dim3 ga(512, 12);
    attn_kernel<<<ga, 128, 0, stream>>>(q1b, kb, vb, biasw, o1b);
    attn_kernel<<<ga, 128, 0, stream>>>(q2b, kb, vb, biasw, o2b);
    attn_kernel<<<ga, 128, 0, stream>>>(q3b, kb, vb, biasw, o3b);

    float* out = (float*)d_out;
    const size_t OB = (size_t)32768 * 384;      // elements per branch output
    gemm_kernel<true, false><<<gg, 256, 0, stream>>>(o1b, wp1, bp1, out + 0 * OB, 1.0f);
    gemm_kernel<true, false><<<gg, 256, 0, stream>>>(o2b, wp2, bp2, out + 1 * OB, 1.0f);
    gemm_kernel<true, false><<<gg, 256, 0, stream>>>(o3b, wp3, bp3, out + 2 * OB, 1.0f);
}
